// Model_Recursive_LSTM_v2_31971736551579
// MI455X (gfx1250) — matrix operand order repaired and checked by reading, not compiled, not run on hardware
//
#include <hip/hip_runtime.h>
#include <hip/hip_bf16.h>
#include <hip/hip_fp16.h>

// ---------------------------------------------------------------------------
// CDNA5 (gfx1250) Model_Recursive_LSTM_v2 forward pass.
// - All GEMMs via v_wmma_f32_16x16x32_f16, fp32 accumulation.
// - Weights pre-converted once per launch to fp16 with K padded to mult. of 32
//   (zero fill) -> B-side tile loads are unconditional global_load_b128.
// - LDS double buffering: one barrier per K-step, next tile's global loads
//   overlap current tile's WMMAs.
// ---------------------------------------------------------------------------

typedef __attribute__((ext_vector_type(16))) _Float16 v16h;
typedef __attribute__((ext_vector_type(8)))  _Float16 v8h;
typedef __attribute__((ext_vector_type(8)))  float    v8f;

#define TM 64
#define TN 64
#define TK 32
#define LDSP 40   // LDS row pitch in halfs (80B rows, 16B-aligned, de-conflicted)

// C[M,N] = act( A[M,K] * W[N,Kp]^T + bias + bias2 (+ C if ACC) )
// W is fp16 with row pitch ldw = Kp = ceil32(K), zero padded.
// ACT: 0 = none, 1 = ELU, 2 = sigmoid
template<int ACT, bool ACC>
__global__ __launch_bounds__(256) void gemm_wmma_k(
    const float* __restrict__ A, int lda,
    const _Float16* __restrict__ W, int ldw,
    const float* __restrict__ bias, const float* __restrict__ bias2,
    float* __restrict__ C, int ldc, int M, int N, int K)
{
  __shared__ _Float16 As[2][TM * LDSP];
  __shared__ _Float16 Bs[2][TN * LDSP];
  const int tid  = threadIdx.x;
  const int lane = tid & 31;
  const int wid  = tid >> 5;
  const int m0 = blockIdx.y * TM;
  const int n0 = blockIdx.x * TN;
  const int wr = (wid & 3) * 16;   // wave row offset in tile
  const int wc = (wid >> 2) * 32;  // wave col offset (2 WMMA cols)
  // loader mapping: 256 threads cover 64 rows x 32 k, 8 consecutive k each
  const int lr = tid >> 2;
  const int lk = (tid & 3) * 8;
  const int gm = m0 + lr;
  const int gn = n0 + lr;
  const long long arow = (long long)((gm < M) ? gm : (M - 1)) * lda;
  const long long wrow = (long long)((gn < N) ? gn : (N - 1)) * ldw;
  const int nt = (K + TK - 1) / TK;

  float av[8]; v8h wv;
  auto load_tile = [&](int ti, float (&a)[8], v8h& w) {
    int kb = ti * TK + lk;
#pragma unroll
    for (int i = 0; i < 8; ++i) {
      int gk = kb + i;
      int ck = (gk < K) ? gk : (K - 1);   // clamp addr, mask value
      float x = A[arow + ck];
      a[i] = (gk < K) ? x : 0.f;
    }
    w = *(const v8h*)(W + wrow + kb);     // padded fp16: always in-bounds
  };
  auto store_tile = [&](int s, const float (&a)[8], v8h w) {
    v8h ah;
#pragma unroll
    for (int i = 0; i < 8; ++i) ah[i] = (_Float16)a[i];
    *(v8h*)&As[s][lr * LDSP + lk] = ah;
    *(v8h*)&Bs[s][lr * LDSP + lk] = w;
  };

  v8f acc0 = {}, acc1 = {};
  load_tile(0, av, wv);
  store_tile(0, av, wv);
  __syncthreads();

  const int g  = lane >> 4;
  const int rr = lane & 15;
  for (int ti = 0; ti < nt; ++ti) {
    const int s = ti & 1;
    const bool more = (ti + 1 < nt);
    float av2[8]; v8h wv2;
    if (more) load_tile(ti + 1, av2, wv2);   // overlap with compute below
    // Fragments per ISA 7.12.2: 16-bit A 16x32 -> lane group g holds K g*8..+7
    // and 16+g*8..+7; B 32x16 -> same K order as A (g*8..+7, 16+g*8..+7).
    const _Float16* ap  = &As[s][(wr + rr) * LDSP];
    const _Float16* bp0 = &Bs[s][(wc + rr) * LDSP];
    const _Float16* bp1 = &Bs[s][(wc + 16 + rr) * LDSP];
    v16h a, b0, b1;
#pragma unroll
    for (int i = 0; i < 8; ++i) {
      a[i]     = ap[g * 8 + i];
      a[8 + i] = ap[16 + g * 8 + i];
    }
#pragma unroll
    for (int i = 0; i < 8; ++i) { b0[i] = bp0[g * 8 + i]; b0[8 + i] = bp0[16 + g * 8 + i]; b1[i] = bp1[g * 8 + i]; b1[8 + i] = bp1[16 + g * 8 + i]; }
    acc0 = __builtin_amdgcn_wmma_f32_16x16x32_f16(false, a, false, b0, (short)0, acc0, false, false);
    acc1 = __builtin_amdgcn_wmma_f32_16x16x32_f16(false, a, false, b1, (short)0, acc1, false, false);
    if (more) store_tile(s ^ 1, av2, wv2);
    __syncthreads();
  }
  // Epilogue: C/D layout -> VGPR j holds row (j + 8*(lane>>4)), col lane&15
  const int cn0 = n0 + wc + (lane & 15);
  const int cn1 = cn0 + 16;
#pragma unroll
  for (int j = 0; j < 8; ++j) {
    int cm = m0 + wr + g * 8 + j;
    if (cm < M) {
      long long rowo = (long long)cm * ldc;
      if (cn0 < N) {
        float v = acc0[j];
        if (bias)  v += bias[cn0];
        if (bias2) v += bias2[cn0];
        if (ACC)   v += C[rowo + cn0];
        if (ACT == 1) v = v > 0.f ? v : expm1f(v);
        if (ACT == 2) v = 1.f / (1.f + expf(-v));
        C[rowo + cn0] = v;
      }
      if (cn1 < N) {
        float v = acc1[j];
        if (bias)  v += bias[cn1];
        if (bias2) v += bias2[cn1];
        if (ACC)   v += C[rowo + cn1];
        if (ACT == 1) v = v > 0.f ? v : expm1f(v);
        if (ACT == 2) v = 1.f / (1.f + expf(-v));
        C[rowo + cn1] = v;
      }
    }
  }
}

// ------------------------- elementwise helpers -----------------------------

// fp32 [N,K] -> fp16 [N,Kp] zero-padded
__global__ void cvt_pad_k(const float* __restrict__ W, _Float16* __restrict__ Wh,
                          long long N, int K, int Kp) {
  long long i = (long long)blockIdx.x * blockDim.x + threadIdx.x;
  long long tot = N * Kp;
  if (i >= tot) return;
  int c = (int)(i % Kp); long long r = i / Kp;
  Wh[i] = (c < K) ? (_Float16)W[r * K + c] : (_Float16)0.f;
}

__global__ void zero_k(float* __restrict__ p, long long n) {
  long long i = (long long)blockIdx.x * blockDim.x + threadIdx.x;
  if (i < n) p[i] = 0.f;
}

// gates [B,4H] in PyTorch order i,f,g,o ; updates c,h in place
__global__ void lstm_pointwise_k(const float* __restrict__ gates,
                                 float* __restrict__ h, float* __restrict__ c,
                                 int Bn, int Hd) {
  long long i = (long long)blockIdx.x * blockDim.x + threadIdx.x;
  long long n = (long long)Bn * Hd;
  if (i >= n) return;
  int b = (int)(i / Hd), k = (int)(i % Hd);
  const float* gp = gates + (long long)b * 4 * Hd;
  float gi = gp[k], gf = gp[Hd + k], gg = gp[2 * Hd + k], go = gp[3 * Hd + k];
  float si = 1.f / (1.f + expf(-gi));
  float sf = 1.f / (1.f + expf(-gf));
  float tg = tanhf(gg);
  float so = 1.f / (1.f + expf(-go));
  float cc = sf * c[i] + si * tg;
  c[i] = cc;
  h[i] = so * tanhf(cc);
}

// out[r, 0..wa-1] = a[r,:], out[r, wa..wa+wb-1] = b[r,:]
__global__ void concat2_k(const float* __restrict__ a, int wa,
                          const float* __restrict__ b, int wb,
                          float* __restrict__ o, long long rows) {
  long long i = (long long)blockIdx.x * blockDim.x + threadIdx.x;
  int wt = wa + wb;
  long long tot = rows * wt;
  if (i >= tot) return;
  long long r = i / wt; int cc = (int)(i % wt);
  o[i] = (cc < wa) ? a[r * wa + cc] : b[r * wb + (cc - wa)];
}

// seq[((s*L + j)*B + b)*Hd + k] = emb[(b*NC + idx[j*S + s])*Hd + k]
__global__ void gather_seq_k(const float* __restrict__ emb,
                             const int* __restrict__ idx,
                             float* __restrict__ seq,
                             int Bn, int NC, int Hd, int L, int S) {
  long long i = (long long)blockIdx.x * blockDim.x + threadIdx.x;
  long long tot = (long long)S * L * Bn * Hd;
  if (i >= tot) return;
  int k = (int)(i % Hd); long long t = i / Hd;
  int b = (int)(t % Bn); t /= Bn;
  int j = (int)(t % L);  int s = (int)(t / L);
  int ci = idx[j * S + s];
  seq[i] = emb[((long long)b * NC + ci) * Hd + k];
}

// seq[((s*Nn + n)*B + b)*Hd + k] = child[((2n+s)*B + b)*Hd + k]
__global__ void pair_seq_k(const float* __restrict__ child,
                           float* __restrict__ seq, int Nn, int Bn, int Hd) {
  long long i = (long long)blockIdx.x * blockDim.x + threadIdx.x;
  long long tot = (long long)2 * Nn * Bn * Hd;
  if (i >= tot) return;
  int k = (int)(i % Hd); long long t = i / Hd;
  int b = (int)(t % Bn); t /= Bn;
  int n = (int)(t % Nn); int s = (int)(t / Nn);
  seq[i] = child[(((long long)(2 * n + s)) * Bn + b) * Hd + k];
}

// X[(n*B+b), :] = [nodes_h | comps_h | loops[b, loop_base+n, 0:2]]
__global__ void build_concat_k(const float* __restrict__ nodes_h,
                               const float* __restrict__ no_nodes,
                               const float* __restrict__ comps_h,
                               const float* __restrict__ no_comps,
                               const float* __restrict__ loops,
                               int loop_base, int Nn, int Bn, int Hd,
                               float* __restrict__ X) {
  int Wd = 2 * Hd + 2;
  long long i = (long long)blockIdx.x * blockDim.x + threadIdx.x;
  long long tot = (long long)Nn * Bn * Wd;
  if (i >= tot) return;
  int cc = (int)(i % Wd); long long r = i / Wd;
  int b = (int)(r % Bn); int n = (int)(r / Bn);
  float v;
  if (cc < Hd)            v = nodes_h ? nodes_h[r * Hd + cc] : no_nodes[cc];
  else if (cc < 2 * Hd) { int k = cc - Hd;
                          v = comps_h ? comps_h[r * Hd + k] : no_comps[k]; }
  else                    v = loops[((long long)b * 14 + loop_base + n) * 2 + (cc - 2 * Hd)];
  X[i] = v;
}

// ------------------------------ host side ----------------------------------

static void launch_gemm(hipStream_t st, const float* A, int lda,
                        const _Float16* W, int ldw,
                        const float* b1, const float* b2,
                        float* C, int ldc, int M, int N, int K,
                        int act, bool accum) {
  dim3 g((N + TN - 1) / TN, (M + TM - 1) / TM), blk(256);
  if (accum)       gemm_wmma_k<0, true ><<<g, blk, 0, st>>>(A, lda, W, ldw, b1, b2, C, ldc, M, N, K);
  else if (act==1) gemm_wmma_k<1, false><<<g, blk, 0, st>>>(A, lda, W, ldw, b1, b2, C, ldc, M, N, K);
  else if (act==2) gemm_wmma_k<2, false><<<g, blk, 0, st>>>(A, lda, W, ldw, b1, b2, C, ldc, M, N, K);
  else             gemm_wmma_k<0, false><<<g, blk, 0, st>>>(A, lda, W, ldw, b1, b2, C, ldc, M, N, K);
}

static void run_lstm(hipStream_t st, const float* x, int lda, long long step_stride,
                     int T, int Bn, int din, int hd,
                     const _Float16* Wih, int kpih,
                     const _Float16* Whh, int kphh,
                     const float* bih, const float* bhh,
                     float* gates, float* h, float* c) {
  long long n = (long long)Bn * hd;
  int zb = (int)((n + 255) / 256);
  zero_k<<<zb, 256, 0, st>>>(h, n);
  zero_k<<<zb, 256, 0, st>>>(c, n);
  for (int t = 0; t < T; ++t) {
    launch_gemm(st, x + (long long)t * step_stride, lda, Wih, kpih, bih, bhh,
                gates, 4 * hd, Bn, 4 * hd, din, 0, false);
    launch_gemm(st, h, hd, Whh, kphh, nullptr, nullptr,
                gates, 4 * hd, Bn, 4 * hd, hd, 0, true);
    lstm_pointwise_k<<<zb, 256, 0, st>>>(gates, h, c, Bn, hd);
  }
}

#define EWL(kern, n, ...) do { long long _n=(n); \
    kern<<<(unsigned)((_n+255)/256),256,0,stream>>>(__VA_ARGS__); } while(0)

extern "C" void kernel_launch(void* const* d_in, const int* in_sizes, int n_in,
                              void* d_out, int out_size, void* d_ws, size_t ws_size,
                              hipStream_t stream) {
  const int B = 128, NC = 64, T = 32, V = 11, F = 846, H = 180, EH = 100;
  const int BN = B * NC; // 8192

  auto P = [&](int i) { return (const float*)d_in[i]; };
  const float* comps = P(0);                 // [128,64,846]
  const float* loops = P(1);                 // [128,14,2]
  const float* expr  = P(2);                 // [128,64,32,11]
  const int*   cidx  = (const int*)d_in[3];  // [8,8]
  // params flat order:
  // 4..11 comp_fc (W,b)x4 | 12..15 concat | 16..19 reg | 20..21 predict
  // 22 no_comps 23 no_nodes | 24..27 comps_lstm | 28..31 nodes_lstm
  // 32..35 roots_lstm | 36..39 expr_lstm  (each lstm: Wih,Whh,bih,bhh)

  float* ws = (float*)d_ws;
  size_t off = 0;
  auto alloc = [&](size_t n) {
    off = (off + 7) & ~(size_t)7;            // keep 32B alignment
    float* p = ws + off; off += n; return p;
  };
  auto allocH = [&](size_t n) { return (_Float16*)alloc((n + 1) / 2); };

  // ---------- Weight pre-conversion: fp32 [N,K] -> fp16 [N,ceil32(K)] ------
  auto kp32 = [](int K) { return (K + 31) & ~31; };
  struct WX { int idx, N, K; };
  const WX wlist[] = {
    {4,600,F+EH},{6,350,600},{8,200,350},{10,H,200},
    {12,200,2*H+2},{14,H,200},{16,200,H},{18,H,200},{20,3,H},
    {24,4*H,H},{25,4*H,H},{28,4*H,H},{29,4*H,H},{32,4*H,H},{33,4*H,H},
    {36,4*EH,V},{37,4*EH,EH}
  };
  _Float16* wh[40] = {};
  int wkp[40] = {};
  for (const WX& w : wlist) {
    int Kp = kp32(w.K);
    wh[w.idx]  = allocH((size_t)w.N * Kp);
    wkp[w.idx] = Kp;
    long long tot = (long long)w.N * Kp;
    cvt_pad_k<<<(unsigned)((tot + 255) / 256), 256, 0, stream>>>(P(w.idx), wh[w.idx], w.N, w.K, Kp);
  }

  // Persistent activations
  float* H_EXPR = alloc((size_t)BN * EH);   // expr_lstm final h [8192,100]
  float* C_EXPR = alloc((size_t)BN * EH);
  float* EMB    = alloc((size_t)BN * H);    // comps_embeddings [8192,180]
  float* pool   = ws + ((off + 7) & ~(size_t)7);  // reused across phases
  // Peak usage ~20.5M floats (~82 MB).

  // ---------- Phase 1: expr LSTM over all B*NC sequences -------------------
  {
    float* GATES = pool;                     // [8192, 400]
    run_lstm(stream, expr, T * V, (long long)V, T, BN, V, EH,
             wh[36], wkp[36], wh[37], wkp[37], P(38), P(39),
             GATES, H_EXPR, C_EXPR);
  }

  // ---------- Phase 2: comp_fc MLP -----------------------------------------
  {
    float* XCAT = pool;                            // [8192, 946]
    float* Y1   = XCAT + (size_t)BN * (F + EH);    // [8192, 600]
    float* Y2   = Y1   + (size_t)BN * 600;         // [8192, 350]
    float* Y3   = Y2   + (size_t)BN * 350;         // [8192, 200]
    EWL(concat2_k, (long long)BN * (F + EH), comps, F, H_EXPR, EH, XCAT, (long long)BN);
    launch_gemm(stream, XCAT, F + EH, wh[4], wkp[4], P(5), nullptr, Y1, 600, BN, 600, F + EH, 1, false);
    launch_gemm(stream, Y1, 600, wh[6], wkp[6], P(7), nullptr, Y2, 350, BN, 350, 600, 1, false);
    launch_gemm(stream, Y2, 350, wh[8], wkp[8], P(9), nullptr, Y3, 200, BN, 200, 350, 1, false);
    launch_gemm(stream, Y3, 200, wh[10], wkp[10], P(11), nullptr, EMB, H, BN, 180, 200, 1, false);
  }

  // ---------- Phase 3: tree (leaves batched 1024, mids 512, roots 256) -----
  float* SEQ   = pool;                         // [8][1024][180]
  float* GATES = SEQ   + (size_t)8 * 1024 * H; // [1024, 720] (max, reused)
  float* H1 = GATES + (size_t)1024 * 4 * H;    float* Cst1 = H1 + (size_t)1024 * H;
  float* X1 = Cst1 + (size_t)1024 * H;         // [1024, 362]
  float* Y1n = X1 + (size_t)1024 * (2 * H + 2);// [1024, 200]
  float* LEAFH = Y1n + (size_t)1024 * 200;     // [1024, 180]
  float* SEQ2 = LEAFH + (size_t)1024 * H;      // [2][512][180]
  float* H2 = SEQ2 + (size_t)2 * 512 * H;      float* Cst2 = H2 + (size_t)512 * H;
  float* X2 = Cst2 + (size_t)512 * H;          float* Y2n = X2 + (size_t)512 * (2 * H + 2);
  float* MIDH = Y2n + (size_t)512 * 200;       // [512, 180]
  float* SEQ3 = MIDH + (size_t)512 * H;        // [2][256][180]
  float* H3 = SEQ3 + (size_t)2 * 256 * H;      float* Cst3 = H3 + (size_t)256 * H;
  float* X3 = Cst3 + (size_t)256 * H;          float* Y3n = X3 + (size_t)256 * (2 * H + 2);
  float* ROOTH = Y3n + (size_t)256 * 200;      // [256, 180]
  float* SEQ4 = ROOTH + (size_t)256 * H;       // [2][128][180]
  float* H4 = SEQ4 + (size_t)2 * B * H;        float* Cst4 = H4 + (size_t)B * H;
  float* R1 = Cst4 + (size_t)B * H;            // [128, 200]
  float* R2 = R1 + (size_t)B * 200;            // [128, 180]

  // Leaves: comps_lstm over gathered index sequences (8 leaves x 128 batch)
  EWL(gather_seq_k, (long long)8 * 8 * B * H, EMB, cidx, SEQ, B, NC, H, 8, 8);
  run_lstm(stream, SEQ, H, (long long)1024 * H, 8, 1024, H, H,
           wh[24], wkp[24], wh[25], wkp[25], P(26), P(27), GATES, H1, Cst1);
  EWL(build_concat_k, (long long)1024 * (2 * H + 2),
      (const float*)nullptr, P(23), H1, P(22), loops, 6, 8, B, H, X1);
  launch_gemm(stream, X1, 2 * H + 2, wh[12], wkp[12], P(13), nullptr, Y1n, 200, 1024, 200, 2 * H + 2, 1, false);
  launch_gemm(stream, Y1n, 200, wh[14], wkp[14], P(15), nullptr, LEAFH, H, 1024, H, 200, 1, false);

  // Mids: nodes_lstm over child pairs (4 nodes x 128)
  EWL(pair_seq_k, (long long)2 * 512 * H, LEAFH, SEQ2, 4, B, H);
  run_lstm(stream, SEQ2, H, (long long)512 * H, 2, 512, H, H,
           wh[28], wkp[28], wh[29], wkp[29], P(30), P(31), GATES, H2, Cst2);
  EWL(build_concat_k, (long long)512 * (2 * H + 2),
      H2, (const float*)nullptr, (const float*)nullptr, P(22), loops, 2, 4, B, H, X2);
  launch_gemm(stream, X2, 2 * H + 2, wh[12], wkp[12], P(13), nullptr, Y2n, 200, 512, 200, 2 * H + 2, 1, false);
  launch_gemm(stream, Y2n, 200, wh[14], wkp[14], P(15), nullptr, MIDH, H, 512, H, 200, 1, false);

  // Roots: nodes_lstm over mid pairs (2 nodes x 128)
  EWL(pair_seq_k, (long long)2 * 256 * H, MIDH, SEQ3, 2, B, H);
  run_lstm(stream, SEQ3, H, (long long)256 * H, 2, 256, H, H,
           wh[28], wkp[28], wh[29], wkp[29], P(30), P(31), GATES, H3, Cst3);
  EWL(build_concat_k, (long long)256 * (2 * H + 2),
      H3, (const float*)nullptr, (const float*)nullptr, P(22), loops, 0, 2, B, H, X3);
  launch_gemm(stream, X3, 2 * H + 2, wh[12], wkp[12], P(13), nullptr, Y3n, 200, 256, 200, 2 * H + 2, 1, false);
  launch_gemm(stream, Y3n, 200, wh[14], wkp[14], P(15), nullptr, ROOTH, H, 256, H, 200, 1, false);

  // roots_lstm over the two root hidden states, then reg MLP + predict
  EWL(pair_seq_k, (long long)2 * B * H, ROOTH, SEQ4, 1, B, H);
  run_lstm(stream, SEQ4, H, (long long)B * H, 2, B, H, H,
           wh[32], wkp[32], wh[33], wkp[33], P(34), P(35), GATES, H4, Cst4);
  launch_gemm(stream, H4, H, wh[16], wkp[16], P(17), nullptr, R1, 200, B, 200, H, 1, false);
  launch_gemm(stream, R1, 200, wh[18], wkp[18], P(19), nullptr, R2, H, B, H, 200, 1, false);
  launch_gemm(stream, R2, H, wh[20], wkp[20], P(21), nullptr, (float*)d_out, 3, B, 3, H, 2, false);
}
